// GraphNet_88407606821031
// MI455X (gfx1250) — matrix operand order repaired and checked by reading, not compiled, not run on hardware
//
#include <hip/hip_runtime.h>
#include <hip/hip_bf16.h>
#include <math.h>

// ---------------------------------------------------------------------------
// GraphNet (EdgeConv x2 + edge classifier) for gfx1250 (MI455X).
// Edge MLP runs on v_wmma_f32_16x16x32_bf16; node table H (12.8MB bf16) is
// L2-resident so per-edge gathers/atomics stay on-chip. Each wave owns two
// 16-edge M-tiles so every B-fragment LDS load feeds two WMMAs.
// ---------------------------------------------------------------------------

#define HID   64
#define HID2  128

typedef __bf16 bf16;
typedef __attribute__((ext_vector_type(16))) __bf16 v16bf;
typedef __attribute__((ext_vector_type(8)))  __bf16 v8bf;
typedef __attribute__((ext_vector_type(8)))  float  v8f;

union V16U { v16bf v; v8bf h[2]; };

// A-fragment (16x32 bf16, M=rows=edges): lane L(0..15) holds row L,
// K = kbase+{0..7} (v0..3) and kbase+16+{0..7} (v4..7); lanes 16..31 hold the
// +8 K-window of the same rows.
__device__ __forceinline__ v16bf frag_A(const bf16* tile, int lane, int kbase) {
    const int r = lane & 15, hh = (lane >> 4) & 1;
    const bf16* p = tile + r * HID2 + kbase + hh * 8;
    V16U u;
    u.h[0] = *(const v8bf*)(p);
    u.h[1] = *(const v8bf*)(p + 16);
    return u.v;
}

// B-fragment (32x16 bf16, N=cols): lane L(0..15) holds column L, K=kbase+{0..7}
// and kbase+16+{0..7}; lanes 16..31 hold the +8 K-window. Wt stored [N][K].
__device__ __forceinline__ v16bf frag_B(const bf16* Wt, int lane, int n0, int kbase) {
    const int r = lane & 15, hh = (lane >> 4) & 1;
    const bf16* p = Wt + (size_t)(n0 + r) * HID2 + kbase + hh * 8;
    V16U u;
    u.h[0] = *(const v8bf*)(p);
    u.h[1] = *(const v8bf*)(p + 16);
    return u.v;
}

// ---------------------------------------------------------------------------
// Weight prep: f32 [K][N] -> bf16 [N][K] (W1t: 128x128, W2t: 64x128), packed
// contiguously so the edgeconv kernel stages both with one LDS copy loop.
// ---------------------------------------------------------------------------
__global__ void prep_weights(const float* __restrict__ W1,
                             const float* __restrict__ W2,
                             bf16* __restrict__ Wt) {
    const int i = blockIdx.x * blockDim.x + threadIdx.x;
    if (i < 16384) {                 // W1t[n*128+k] = W1[k*128+n]
        const int n = i >> 7, k = i & 127;
        Wt[i] = (bf16)W1[k * HID2 + n];
    }
    if (i < 8192) {                  // W2t[n*128+k] = W2[k*64+n]
        const int n = i >> 7, k = i & 127;
        Wt[16384 + i] = (bf16)W2[k * HID + n];
    }
}

// ---------------------------------------------------------------------------
// Input network: H = tanh(BN(x @ W_in + b_in)) -> bf16 [N][64]
// ---------------------------------------------------------------------------
__global__ void input_net_kernel(const float* __restrict__ x,
                                 const float* __restrict__ W_in,
                                 const float* __restrict__ b_in,
                                 const float* __restrict__ gamma,
                                 const float* __restrict__ beta,
                                 const float* __restrict__ mean,
                                 const float* __restrict__ var,
                                 bf16* __restrict__ H, int nNodes) {
    const int n = blockIdx.x * blockDim.x + threadIdx.x;
    if (n >= nNodes) return;
    float xv[8];
#pragma unroll
    for (int k = 0; k < 8; ++k) xv[k] = x[(size_t)n * 8 + k];
    for (int j = 0; j < HID; ++j) {
        float acc = b_in[j];
#pragma unroll
        for (int k = 0; k < 8; ++k) acc += xv[k] * W_in[k * HID + j];
        const float s = rsqrtf(var[j] + 1e-5f);
        acc = (acc - mean[j]) * s * gamma[j] + beta[j];
        H[(size_t)n * HID + j] = (bf16)tanhf(acc);
    }
}

__global__ void zero_f32(float* __restrict__ p, int n) {
    const int i = blockIdx.x * blockDim.x + threadIdx.x;
    if (i < n) p[i] = 0.0f;
}

// re-quantize accumulator to bf16 for next layer, and re-zero the accumulator
__global__ void cvt_zero(float* __restrict__ acc, bf16* __restrict__ Hb, int n) {
    const int i = blockIdx.x * blockDim.x + threadIdx.x;
    if (i < n) { Hb[i] = (bf16)acc[i]; acc[i] = 0.0f; }
}

// ---------------------------------------------------------------------------
// EdgeConv layer. 4 waves/block, 32 edges/wave (two 16-row M-tiles):
//   A = [x_i | x_j - x_i]  (32x128 bf16, gathered into LDS with b128 accesses)
//   M1 = relu(A @ W1 + b1) (32x128: 64 WMMAs, B-fragment shared by 2 M-tiles)
//   M2 = relu(M1 @ W2 + b2)(32x64 : 32 WMMAs)
//   Hacc[dst] += M2        (segment_sum via global f32 atomics)
// ---------------------------------------------------------------------------
__global__ void edgeconv_kernel(const bf16*  __restrict__ Hin,
                                float*       __restrict__ Hacc,
                                const int*   __restrict__ edge_index,
                                const float* __restrict__ b1,
                                const float* __restrict__ b2,
                                const bf16*  __restrict__ Wt,
                                int nEdges) {
    extern __shared__ char smem_raw[];
    bf16* Ws    = (bf16*)smem_raw;          // 24576 bf16: W1t(16384) + W2t(8192)
    bf16* tiles = Ws + 24576;               // 4 waves * (A 4096 + M1 4096)
    int*  ids   = (int*)(tiles + 4 * 8192); // 4 waves * (src[32] + dst[32])

    const int tid  = threadIdx.x;
    const int wave = tid >> 5;
    const int lane = tid & 31;
    const int r    = lane & 15;
    const int hh   = (lane >> 4) & 1;

    // stage both weight matrices to LDS (48KB) with b128 copies
    {
        const uint4* s = (const uint4*)Wt;
        uint4*       d = (uint4*)Ws;
        for (int i = tid; i < 3072; i += blockDim.x) d[i] = s[i];
    }
    __syncthreads();

    bf16* A   = tiles + wave * 8192;   // [32][128]
    bf16* M1  = A + 4096;              // [32][128]
    int*  myS = ids + wave * 64;
    int*  myD = myS + 32;

    const int tileBase = (blockIdx.x * 4 + wave) * 32;

    // ---- stage this wave's 32 edge (src,dst) node ids (clamped) ----
    {
        const int eidx = tileBase + lane;
        const int ec   = eidx < nEdges ? eidx : nEdges - 1;
        myS[lane] = edge_index[ec];            // src
        myD[lane] = edge_index[nEdges + ec];   // dst
    }

    // ---- gather: A[e] = [H[dst] | H[src]-H[dst]] as (edge,chunk) b128 ops ----
    // 32 edges x 8 chunks of 8 bf16 = 256 items; 8 iterations over 32 lanes.
#pragma unroll
    for (int it = 0; it < 8; ++it) {
        const int pi = it * 32 + lane;  // 0..255
        const int e  = pi >> 3;         // edge slot 0..31
        const int c  = (pi & 7) * 8;    // feature base 0..56
        const int sN = myS[e], dN = myD[e];
        const v8bf hi = *(const v8bf*)(Hin + (size_t)dN * HID + c);
        const v8bf hj = *(const v8bf*)(Hin + (size_t)sN * HID + c);
        *(v8bf*)(A + e * HID2 + c) = hi;
        v8bf df;
#pragma unroll
        for (int t = 0; t < 8; ++t) df[t] = (bf16)((float)hj[t] - (float)hi[t]);
        *(v8bf*)(A + e * HID2 + HID + c) = df;
    }
    __syncthreads();

    const bf16* W1s = Ws;          // [128][128]
    const bf16* W2s = Ws + 16384;  // [64][128]

    // ---- layer 1: [32x128] = A @ W1, relu -> M1 (LDS) ----
#pragma unroll
    for (int nt = 0; nt < 8; ++nt) {
        v8f acc0 = {}, acc1 = {};
#pragma unroll
        for (int kk = 0; kk < 4; ++kk) {
            const v16bf b  = frag_B(W1s, lane, nt * 16, kk * 32);   // shared
            const v16bf a0 = frag_A(A,              lane, kk * 32);
            const v16bf a1 = frag_A(A + 16 * HID2,  lane, kk * 32);
            acc0 = __builtin_amdgcn_wmma_f32_16x16x32_bf16(
                false, a0, false, b, (short)0, acc0, false, false);
            acc1 = __builtin_amdgcn_wmma_f32_16x16x32_bf16(
                false, a1, false, b, (short)0, acc1, false, false);
        }
        const float bias = b1[nt * 16 + r];
#pragma unroll
        for (int g = 0; g < 8; ++g) {   // C: row = g + 8*hh, col = r
            const float v0 = fmaxf(acc0[g] + bias, 0.0f);
            const float v1 = fmaxf(acc1[g] + bias, 0.0f);
            M1[(g + hh * 8) * HID2 + nt * 16 + r]        = (bf16)v0;
            M1[(16 + g + hh * 8) * HID2 + nt * 16 + r]   = (bf16)v1;
        }
    }
    __syncthreads();

    // ---- layer 2: [32x64] = M1 @ W2, relu, atomic scatter ----
#pragma unroll
    for (int nt = 0; nt < 4; ++nt) {
        v8f acc0 = {}, acc1 = {};
#pragma unroll
        for (int kk = 0; kk < 4; ++kk) {
            const v16bf b  = frag_B(W2s, lane, nt * 16, kk * 32);   // shared
            const v16bf a0 = frag_A(M1,             lane, kk * 32);
            const v16bf a1 = frag_A(M1 + 16 * HID2, lane, kk * 32);
            acc0 = __builtin_amdgcn_wmma_f32_16x16x32_bf16(
                false, a0, false, b, (short)0, acc0, false, false);
            acc1 = __builtin_amdgcn_wmma_f32_16x16x32_bf16(
                false, a1, false, b, (short)0, acc1, false, false);
        }
        const float bias = b2[nt * 16 + r];
#pragma unroll
        for (int g = 0; g < 8; ++g) {
            const int m0 = g + hh * 8;
            const int m1 = 16 + m0;
            const float v0 = fmaxf(acc0[g] + bias, 0.0f);
            const float v1 = fmaxf(acc1[g] + bias, 0.0f);
            if (tileBase + m0 < nEdges)
                atomicAdd(Hacc + (size_t)myD[m0] * HID + nt * 16 + r, v0);
            if (tileBase + m1 < nEdges)
                atomicAdd(Hacc + (size_t)myD[m1] * HID + nt * 16 + r, v1);
        }
    }
}

// ---------------------------------------------------------------------------
// Edge classifier: sigmoid(dot(H[src], We[0:64]) + dot(H[dst], We[64:128]) + be)
// ---------------------------------------------------------------------------
__global__ void edge_cls(const float* __restrict__ Hf,
                         const int*   __restrict__ edge_index,
                         const float* __restrict__ We,
                         const float* __restrict__ be,
                         float* __restrict__ out, int nEdges) {
    const int e = blockIdx.x * blockDim.x + threadIdx.x;
    if (e >= nEdges) return;
    const int s = edge_index[e], d = edge_index[nEdges + e];
    const float4* hs = (const float4*)(Hf + (size_t)s * HID);
    const float4* hd = (const float4*)(Hf + (size_t)d * HID);
    float z = be[0];
#pragma unroll
    for (int j = 0; j < HID / 4; ++j) {
        const float4 a = hs[j];
        z += a.x * We[4 * j + 0] + a.y * We[4 * j + 1] +
             a.z * We[4 * j + 2] + a.w * We[4 * j + 3];
    }
#pragma unroll
    for (int j = 0; j < HID / 4; ++j) {
        const float4 a = hd[j];
        z += a.x * We[HID + 4 * j + 0] + a.y * We[HID + 4 * j + 1] +
             a.z * We[HID + 4 * j + 2] + a.w * We[HID + 4 * j + 3];
    }
    out[e] = 1.0f / (1.0f + expf(-z));
}

// ---------------------------------------------------------------------------
extern "C" void kernel_launch(void* const* d_in, const int* in_sizes, int n_in,
                              void* d_out, int out_size, void* d_ws, size_t ws_size,
                              hipStream_t stream) {
    const float* x        = (const float*)d_in[0];
    const int*   eidx     = (const int*)  d_in[1];
    const float* W_in     = (const float*)d_in[2];
    const float* b_in     = (const float*)d_in[3];
    const float* bn_gamma = (const float*)d_in[4];
    const float* bn_beta  = (const float*)d_in[5];
    const float* bn_mean  = (const float*)d_in[6];
    const float* bn_var   = (const float*)d_in[7];
    const float* W1       = (const float*)d_in[8];
    const float* b1       = (const float*)d_in[9];
    const float* W2       = (const float*)d_in[10];
    const float* b2       = (const float*)d_in[11];
    const float* We       = (const float*)d_in[12];
    const float* be       = (const float*)d_in[13];

    const int N = in_sizes[0] / 8;   // 100000
    const int E = in_sizes[1] / 2;   // 1600000
    const int NH = N * HID;          // 6.4M elements

    // workspace layout (256B-aligned sections)
    char* ws = (char*)d_ws;
    size_t off = 0;
    bf16*  Hb   = (bf16*)(ws + off);  off += ((size_t)NH * sizeof(bf16) + 255) & ~(size_t)255;
    float* Hacc = (float*)(ws + off); off += ((size_t)NH * sizeof(float) + 255) & ~(size_t)255;
    bf16*  Wt   = (bf16*)(ws + off);  off += (24576 * sizeof(bf16) + 255) & ~(size_t)255;

    const int SMEM = 24576 * 2 + 4 * 8192 * 2 + 4 * 64 * 4;  // 115712 B
    (void)hipFuncSetAttribute((const void*)edgeconv_kernel,
                              hipFuncAttributeMaxDynamicSharedMemorySize, SMEM);

    prep_weights<<<(16384 + 255) / 256, 256, 0, stream>>>(W1, W2, Wt);
    input_net_kernel<<<(N + 255) / 256, 256, 0, stream>>>(
        x, W_in, b_in, bn_gamma, bn_beta, bn_mean, bn_var, Hb, N);
    zero_f32<<<(NH + 255) / 256, 256, 0, stream>>>(Hacc, NH);

    const int ecBlocks = (E + 127) / 128;  // 4 waves * 32 edges per block
    edgeconv_kernel<<<ecBlocks, 128, SMEM, stream>>>(Hb, Hacc, eidx, b1, b2, Wt, E);
    cvt_zero<<<(NH + 255) / 256, 256, 0, stream>>>(Hacc, Hb, NH);
    edgeconv_kernel<<<ecBlocks, 128, SMEM, stream>>>(Hb, Hacc, eidx, b1, b2, Wt, E);

    edge_cls<<<(E + 255) / 256, 256, 0, stream>>>(Hacc, eidx, We, be, (float*)d_out, E);
}
